// PointPillars_13546326851956
// MI455X (gfx1250) — matrix operand order repaired and checked by reading, not compiled, not run on hardware
//
#include <hip/hip_runtime.h>
#include <hip/hip_bf16.h>

// ---------------- geometry constants ----------------
static constexpr int NXC  = 432;
static constexpr int NYC  = 496;
static constexpr int BC   = 2;
static constexpr int PC   = 24000;
static constexpr int NPTSC = 100;
static constexpr int CPFC = 64;
static constexpr int HOF  = 248;   // upsampled feature map H
static constexpr int WOF  = 216;   // upsampled feature map W
static constexpr int CF   = 384;   // concat channels

typedef __attribute__((ext_vector_type(16))) _Float16 v16h;
typedef __attribute__((ext_vector_type(8)))  float    v8f;

union AFrag { v16h v; uint4 q[2]; };

#define WMMA_F16(a, b, c) \
  __builtin_amdgcn_wmma_f32_16x16x32_f16(false, (a), false, (b), (short)0, (c), false, false)

// ---------------- utility kernels ----------------
__global__ void zero_u32_kernel(unsigned int* __restrict__ p, size_t n) {
  size_t i = (size_t)blockIdx.x * blockDim.x + threadIdx.x;
  if (i < n) p[i] = 0u;
}

// ---------------- PFN: per-pillar 9->64 matmul + BN + ReLU + masked max ----------------
__global__ void pfn_kernel(const float* __restrict__ voxels,
                           const int*   __restrict__ num_points,
                           const int*   __restrict__ coors,
                           const float* __restrict__ pw,   // (9,64)
                           const float* __restrict__ pg,   // (64)
                           const float* __restrict__ pb,   // (64)
                           float* __restrict__ pillar)     // (P,64)
{
  __shared__ float s_mean[3];
  __shared__ float s_feat[9];
  const int p = blockIdx.x;
  const int c = threadIdx.x;  // 0..63 output channel
  int np = num_points[p];
  if (np < 1) np = 1;
  if (np > NPTSC) np = NPTSC;
  const float npc = (float)np;
  const float cx = (float)coors[p*4+3] * 0.16f + 0.08f;
  const float cy = (float)coors[p*4+2] * 0.16f + (-39.6f);

  float wreg[9];
#pragma unroll
  for (int i = 0; i < 9; ++i) wreg[i] = pw[i*CPFC + c];
  const float gc = pg[c], bc = pb[c];

  if (c < 3) {
    float s = 0.f;
    for (int j = 0; j < np; ++j) s += voxels[((size_t)p*NPTSC + j)*4 + c];
    s_mean[c] = s / npc;
  }
  __syncthreads();
  float maxv = 0.f;
  for (int j = 0; j < np; ++j) {
    if (c < 9) {
      const float* vj = voxels + ((size_t)p*NPTSC + j)*4;
      float fv;
      if (c < 4)      fv = vj[c];
      else if (c < 7) fv = vj[c-4] - s_mean[c-4];
      else if (c == 7) fv = vj[0] - cx;
      else             fv = vj[1] - cy;
      s_feat[c] = fv;
    }
    __syncthreads();
    float acc = 0.f;
#pragma unroll
    for (int i = 0; i < 9; ++i) acc += s_feat[i] * wreg[i];
    float hv = fmaxf(acc * gc + bc, 0.f);
    maxv = fmaxf(maxv, hv);
    __syncthreads();
  }
  pillar[(size_t)p*CPFC + c] = maxv;
}

// ---------------- scatter pillars into halo-padded NHWC f16 canvas ----------------
__global__ void scatter_kernel(const float* __restrict__ pillar,
                               const int*   __restrict__ coors,
                               _Float16* __restrict__ canvas) // (B,NY+2,NX+2,64)
{
  const int p = blockIdx.x;
  const int c = threadIdx.x;
  const int b  = coors[p*4+0];
  const int yy = coors[p*4+2];
  const int xx = coors[p*4+3];
  const size_t sidx = ((size_t)b*(NYC+2) + (yy+1))*(NXC+2) + (xx+1);
  canvas[sidx*CPFC + c] = (_Float16)pillar[(size_t)p*CPFC + c];
}

// ---------------- weight packing into WMMA B-fragment order ----------------
// packed layout: [(kc*NT + nt)*512 + lane*16 + h], k = kc*32 + (h<8 ? 8*half+h : 16+8*half+h-8), half = lane>=16,
// n = nt*16 + (lane&15)
__global__ void pack_conv_w(const float* __restrict__ w,  // (O,I,3,3)
                            _Float16* __restrict__ wp, int Cin, int Cout)
{
  const int NT = Cout >> 4;
  const size_t total = (size_t)Cin * 9 * Cout;
  size_t idx = (size_t)blockIdx.x * blockDim.x + threadIdx.x;
  if (idx >= total) return;
  int h    = (int)(idx & 15);
  int lane = (int)((idx >> 4) & 31);
  size_t t = idx >> 9;
  int nt = (int)(t % NT);
  int kc = (int)(t / NT);
  int k = kc*32 + ((lane & 16) ? 8 : 0) + ((h & 8) ? 16 : 0) + (h & 7);   // k = r*Cin + ci
  int n = nt*16 + (lane & 15);
  int r  = k / Cin;
  int ci = k - r*Cin;
  int ky = r / 3, kx = r - ky*3;
  wp[idx] = (_Float16)w[(((size_t)n*Cin + ci)*3 + ky)*3 + kx];
}

__global__ void pack_deconv_w(const float* __restrict__ w,  // (C,128,s,s)
                              _Float16* __restrict__ wp, int Cin, int s)
{
  const size_t per = (size_t)Cin * 128;
  const size_t total = per * s * s;
  size_t idx = (size_t)blockIdx.x * blockDim.x + threadIdx.x;
  if (idx >= total) return;
  int dk = (int)(idx / per);
  size_t rem = idx - (size_t)dk * per;
  int h    = (int)(rem & 15);
  int lane = (int)((rem >> 4) & 31);
  size_t t = rem >> 9;
  int nt = (int)(t & 7);       // NT = 8 (Cout=128)
  int kc = (int)(t >> 3);
  int k = kc*32 + ((lane & 16) ? 8 : 0) + ((h & 8) ? 16 : 0) + (h & 7);   // = input channel c
  int n = nt*16 + (lane & 15);                  // = output channel o
  int d  = dk / s;
  int kk = dk - d*s;
  wp[idx] = (_Float16)w[(((size_t)k*128 + n)*s + d)*s + kk];
}

__global__ void pack_head_w(const float* __restrict__ box_w,  // (14,384)
                            const float* __restrict__ cls_w,  // (2,384)
                            _Float16* __restrict__ wp)        // K=384, N=16
{
  const size_t total = (size_t)CF * 16;
  size_t idx = (size_t)blockIdx.x * blockDim.x + threadIdx.x;
  if (idx >= total) return;
  int h    = (int)(idx & 15);
  int lane = (int)((idx >> 4) & 31);
  int kc   = (int)(idx >> 9);         // NT = 1
  int k = kc*32 + ((lane & 16) ? 8 : 0) + ((h & 8) ? 16 : 0) + (h & 7);
  int n = lane & 15;
  float v = (n < 14) ? box_w[(size_t)n*CF + k] : cls_w[(size_t)(n-14)*CF + k];
  wp[idx] = (_Float16)v;
}

// ---------------- 3x3 conv (pad 1, stride s) as implicit GEMM on WMMA ----------------
// Halo-padded NHWC f16 in/out: in (B,Hi+2,Wi+2,Cin), out (B,Ho+2,Wo+2,Cout);
// borders are pre-zeroed so the inner loop has NO bounds checks.
// Per wave: 32(M pixels) x 64(N ochan) = 8 accumulators; B fragments reused
// across the two M subtiles. fp32 accumulate, fused BN+ReLU.
__global__ void conv3x3_wmma(const _Float16* __restrict__ in,
                             const _Float16* __restrict__ wp,
                             const float* __restrict__ gamma,
                             const float* __restrict__ beta,
                             _Float16* __restrict__ out,
                             int Bn, int Hi, int Wi, int Cin,
                             int Ho, int Wo, int Cout, int stride)
{
  const int lane = threadIdx.x & 31;
  const int wave = threadIdx.x >> 5;
  const int M = Bn * Ho * Wo;
  const int mbase = (blockIdx.x * 4 + wave) * 32;
  if (mbase >= M) return;                 // wave-uniform
  const bool two = (mbase + 16) < M;      // wave-uniform second subtile
  const int NT = Cout >> 4;
  const int CC = Cin >> 5;                // K-chunks per tap
  const int nt0 = blockIdx.y * 4;         // 4 consecutive 16-wide N tiles
  const int Hip = Hi + 2, Wip = Wi + 2, Wop = Wo + 2;

  // A-fragment geometry: lane -> row M=lane&15; K base 0/8 by half-wave
  const int mrow = lane & 15;
  const int kb = (lane & 16) ? 8 : 0;
  const int pix0 = mbase + mrow;
  const int pix1 = two ? (mbase + 16 + mrow) : pix0;

  const int HWo = Ho * Wo;
  const int b0p = pix0 / HWo; const int hw0 = pix0 - b0p * HWo;
  const int ho0 = hw0 / Wo, wo0 = hw0 - (hw0 / Wo) * Wo;
  const int b1p = pix1 / HWo; const int hw1 = pix1 - b1p * HWo;
  const int ho1 = hw1 / Wo, wo1 = hw1 - (hw1 / Wo) * Wo;
  // top-left 3x3 tap in padded coordinates (pad cancels the -1)
  const _Float16* a0base =
      in + (((size_t)b0p * Hip + ho0 * stride) * Wip + wo0 * stride) * Cin + kb;
  const _Float16* a1base =
      in + (((size_t)b1p * Hip + ho1 * stride) * Wip + wo1 * stride) * Cin + kb;

  v8f a0c0 = {}, a0c1 = {}, a0c2 = {}, a0c3 = {};
  v8f a1c0 = {}, a1c1 = {}, a1c2 = {}, a1c3 = {};
  const _Float16* wptr = wp + (size_t)nt0 * 512 + (size_t)lane * 16;
  const size_t wstep = (size_t)NT * 512;

#pragma unroll
  for (int r = 0; r < 9; ++r) {
    const int ky = r / 3, kx = r - (r / 3) * 3;      // constants after unroll
    const _Float16* a0p = a0base + ((size_t)ky * Wip + kx) * Cin;
    const _Float16* a1p = a1base + ((size_t)ky * Wip + kx) * Cin;
    for (int cchunk = 0; cchunk < CC; ++cchunk) {
      AFrag a0, a1;
      a0.q[0] = *(const uint4*)(a0p); a0.q[1] = *(const uint4*)(a0p + 16);
      a1.q[0] = *(const uint4*)(a1p); a1.q[1] = *(const uint4*)(a1p + 16);
      a0p += 32; a1p += 32;
      v16h b0 = *(const v16h*)(wptr);
      v16h b1 = *(const v16h*)(wptr + 512);
      v16h b2 = *(const v16h*)(wptr + 1024);
      v16h b3 = *(const v16h*)(wptr + 1536);
      __builtin_prefetch(wptr + wstep, 0, 1);
      wptr += wstep;
      a0c0 = WMMA_F16(a0.v, b0, a0c0);
      a1c0 = WMMA_F16(a1.v, b0, a1c0);
      a0c1 = WMMA_F16(a0.v, b1, a0c1);
      a1c1 = WMMA_F16(a1.v, b1, a1c1);
      a0c2 = WMMA_F16(a0.v, b2, a0c2);
      a1c2 = WMMA_F16(a1.v, b2, a1c2);
      a0c3 = WMMA_F16(a0.v, b3, a0c3);
      a1c3 = WMMA_F16(a1.v, b3, a1c3);
    }
  }

  // C/D layout: lane col n = lane&15; row M = vgpr + (lane<16 ? 0 : 8)
  const int n = lane & 15;
  const int mo = (lane < 16) ? 0 : 8;
  const int ch0 = nt0 * 16 + n;
  float gr[4], br[4];
#pragma unroll
  for (int g = 0; g < 4; ++g) { gr[g] = gamma[ch0 + g*16]; br[g] = beta[ch0 + g*16]; }
#pragma unroll
  for (int v = 0; v < 8; ++v) {
    const int p2 = mbase + mo + v;
    const int b2 = p2 / HWo; const int hw = p2 - b2 * HWo;
    const int ho = hw / Wo, wo = hw - (hw / Wo) * Wo;
    const size_t ob = (((size_t)b2 * (Ho + 2) + ho + 1) * Wop + wo + 1) * Cout + ch0;
    out[ob     ] = (_Float16)fmaxf(a0c0[v]*gr[0] + br[0], 0.f);
    out[ob + 16] = (_Float16)fmaxf(a0c1[v]*gr[1] + br[1], 0.f);
    out[ob + 32] = (_Float16)fmaxf(a0c2[v]*gr[2] + br[2], 0.f);
    out[ob + 48] = (_Float16)fmaxf(a0c3[v]*gr[3] + br[3], 0.f);
  }
  if (two) {
#pragma unroll
    for (int v = 0; v < 8; ++v) {
      const int p2 = mbase + 16 + mo + v;
      const int b2 = p2 / HWo; const int hw = p2 - b2 * HWo;
      const int ho = hw / Wo, wo = hw - (hw / Wo) * Wo;
      const size_t ob = (((size_t)b2 * (Ho + 2) + ho + 1) * Wop + wo + 1) * Cout + ch0;
      out[ob     ] = (_Float16)fmaxf(a1c0[v]*gr[0] + br[0], 0.f);
      out[ob + 16] = (_Float16)fmaxf(a1c1[v]*gr[1] + br[1], 0.f);
      out[ob + 32] = (_Float16)fmaxf(a1c2[v]*gr[2] + br[2], 0.f);
      out[ob + 48] = (_Float16)fmaxf(a1c3[v]*gr[3] + br[3], 0.f);
    }
  }
}

// ---------------- transposed conv as s*s GEMMs on WMMA ----------------
// Reads halo-padded input (B,Hi+2,Wi+2,Cin); writes unpadded concat buffer.
// out(b, hi*s+d, wi*s+k, chanBase+o) = relu(BN(sum_c x(b,hi,wi,c) * w(c,o,d,k)))
__global__ void deconv_wmma(const _Float16* __restrict__ in,
                            const _Float16* __restrict__ wp,
                            const float* __restrict__ gamma,
                            const float* __restrict__ beta,
                            _Float16* __restrict__ outc,   // (B,HOF,WOF,CF)
                            int Hi, int Wi, int Cin, int s, int chanBase)
{
  const int lane = threadIdx.x & 31;
  const int wave = threadIdx.x >> 5;
  const int M = BC * Hi * Wi;
  const int mbase = (blockIdx.x * 4 + wave) * 32;
  if (mbase >= M) return;
  const bool two = (mbase + 16) < M;
  const int dk = blockIdx.z;
  const int d = dk / s, kk = dk - d * s;
  const int KC = Cin >> 5;
  const int nt0 = blockIdx.y * 4;
  const int Hip = Hi + 2, Wip = Wi + 2;

  const int mrow = lane & 15;
  const int kb = (lane & 16) ? 8 : 0;
  const int pix0 = mbase + mrow;
  const int pix1 = two ? (mbase + 16 + mrow) : pix0;
  const int HWi = Hi * Wi;
  const int b0 = pix0 / HWi; const int hw0 = pix0 - b0 * HWi;
  const int hi0 = hw0 / Wi, wi0 = hw0 - (hw0 / Wi) * Wi;
  const int b1 = pix1 / HWi; const int hw1 = pix1 - b1 * HWi;
  const int hi1 = hw1 / Wi, wi1 = hw1 - (hw1 / Wi) * Wi;
  const _Float16* a0p =
      in + (((size_t)b0 * Hip + hi0 + 1) * Wip + wi0 + 1) * Cin + kb;
  const _Float16* a1p =
      in + (((size_t)b1 * Hip + hi1 + 1) * Wip + wi1 + 1) * Cin + kb;

  v8f a0c0 = {}, a0c1 = {}, a0c2 = {}, a0c3 = {};
  v8f a1c0 = {}, a1c1 = {}, a1c2 = {}, a1c3 = {};
  const _Float16* wptr = wp + (size_t)dk * KC * 8 * 512
                            + (size_t)nt0 * 512 + (size_t)lane * 16;
  for (int kc = 0; kc < KC; ++kc) {
    AFrag a0, a1;
    a0.q[0] = *(const uint4*)(a0p); a0.q[1] = *(const uint4*)(a0p + 16);
    a1.q[0] = *(const uint4*)(a1p); a1.q[1] = *(const uint4*)(a1p + 16);
    a0p += 32; a1p += 32;
    v16h b0f = *(const v16h*)(wptr);
    v16h b1f = *(const v16h*)(wptr + 512);
    v16h b2f = *(const v16h*)(wptr + 1024);
    v16h b3f = *(const v16h*)(wptr + 1536);
    wptr += (size_t)8 * 512;
    a0c0 = WMMA_F16(a0.v, b0f, a0c0);
    a1c0 = WMMA_F16(a1.v, b0f, a1c0);
    a0c1 = WMMA_F16(a0.v, b1f, a0c1);
    a1c1 = WMMA_F16(a1.v, b1f, a1c1);
    a0c2 = WMMA_F16(a0.v, b2f, a0c2);
    a1c2 = WMMA_F16(a1.v, b2f, a1c2);
    a0c3 = WMMA_F16(a0.v, b3f, a0c3);
    a1c3 = WMMA_F16(a1.v, b3f, a1c3);
  }

  const int n = lane & 15;
  const int mo = (lane < 16) ? 0 : 8;
  const int ch0 = chanBase + nt0 * 16 + n;
  float gr[4], br[4];
#pragma unroll
  for (int g = 0; g < 4; ++g) {
    gr[g] = gamma[nt0*16 + n + g*16];
    br[g] = beta [nt0*16 + n + g*16];
  }
#pragma unroll
  for (int v = 0; v < 8; ++v) {
    const int p2 = mbase + mo + v;
    const int b2 = p2 / HWi; const int hw = p2 - b2 * HWi;
    const int hi = hw / Wi, wi = hw - (hw / Wi) * Wi;
    const size_t ob = (((size_t)b2 * HOF + hi*s + d) * WOF + wi*s + kk) * CF + ch0;
    outc[ob     ] = (_Float16)fmaxf(a0c0[v]*gr[0] + br[0], 0.f);
    outc[ob + 16] = (_Float16)fmaxf(a0c1[v]*gr[1] + br[1], 0.f);
    outc[ob + 32] = (_Float16)fmaxf(a0c2[v]*gr[2] + br[2], 0.f);
    outc[ob + 48] = (_Float16)fmaxf(a0c3[v]*gr[3] + br[3], 0.f);
  }
  if (two) {
#pragma unroll
    for (int v = 0; v < 8; ++v) {
      const int p2 = mbase + 16 + mo + v;
      const int b2 = p2 / HWi; const int hw = p2 - b2 * HWi;
      const int hi = hw / Wi, wi = hw - (hw / Wi) * Wi;
      const size_t ob = (((size_t)b2 * HOF + hi*s + d) * WOF + wi*s + kk) * CF + ch0;
      outc[ob     ] = (_Float16)fmaxf(a1c0[v]*gr[0] + br[0], 0.f);
      outc[ob + 16] = (_Float16)fmaxf(a1c1[v]*gr[1] + br[1], 0.f);
      outc[ob + 32] = (_Float16)fmaxf(a1c2[v]*gr[2] + br[2], 0.f);
      outc[ob + 48] = (_Float16)fmaxf(a1c3[v]*gr[3] + br[3], 0.f);
    }
  }
}

// ---------------- detection heads: 384 -> 16 (14 box + 2 cls) GEMM ----------------
__global__ void head_wmma(const _Float16* __restrict__ f,    // (B,HOF,WOF,384)
                          const _Float16* __restrict__ wp,   // packed 384x16
                          const float* __restrict__ box_b,   // (14)
                          const float* __restrict__ cls_b,   // (2)
                          float* __restrict__ out)           // box flat then cls flat
{
  const int lane = threadIdx.x & 31;
  const int wave = threadIdx.x >> 5;
  const int M = BC * HOF * WOF;
  const int mbase = (blockIdx.x * 4 + wave) * 32;
  if (mbase >= M) return;
  const bool two = (mbase + 16) < M;
  const int mrow = lane & 15;
  const int kb = (lane & 16) ? 8 : 0;
  const int pix0 = mbase + mrow;
  const int pix1 = two ? (mbase + 16 + mrow) : pix0;
  const _Float16* a0p = f + (size_t)pix0 * CF + kb;
  const _Float16* a1p = f + (size_t)pix1 * CF + kb;

  v8f acc0 = {}, acc1 = {};
  const _Float16* wptr = wp + (size_t)lane * 16;
  for (int kc = 0; kc < CF / 32; ++kc) {
    AFrag a0, a1;
    a0.q[0] = *(const uint4*)(a0p); a0.q[1] = *(const uint4*)(a0p + 16);
    a1.q[0] = *(const uint4*)(a1p); a1.q[1] = *(const uint4*)(a1p + 16);
    a0p += 32; a1p += 32;
    v16h bfr = *(const v16h*)(wptr);
    wptr += 512;
    acc0 = WMMA_F16(a0.v, bfr, acc0);
    acc1 = WMMA_F16(a1.v, bfr, acc1);
  }

  const int n = lane & 15;
  const int mo = (lane < 16) ? 0 : 8;
  const size_t clsOff = (size_t)BC * HOF * WOF * 14;
#pragma unroll
  for (int v = 0; v < 8; ++v) {
    const int p2 = mbase + mo + v;
    if (n < 14) out[(size_t)p2 * 14 + n] = acc0[v] + box_b[n];
    else        out[clsOff + (size_t)p2 * 2 + (n - 14)] = acc0[v] + cls_b[n - 14];
  }
  if (two) {
#pragma unroll
    for (int v = 0; v < 8; ++v) {
      const int p2 = mbase + 16 + mo + v;
      if (n < 14) out[(size_t)p2 * 14 + n] = acc1[v] + box_b[n];
      else        out[clsOff + (size_t)p2 * 2 + (n - 14)] = acc1[v] + cls_b[n - 14];
    }
  }
}

// ---------------- host-side orchestration ----------------
extern "C" void kernel_launch(void* const* d_in, const int* in_sizes, int n_in,
                              void* d_out, int out_size, void* d_ws, size_t ws_size,
                              hipStream_t stream)
{
  (void)in_sizes; (void)n_in; (void)out_size; (void)ws_size;

  const float* voxels     = (const float*)d_in[0];
  const int*   num_points = (const int*)  d_in[1];
  const int*   coors      = (const int*)  d_in[2];
  const float* pfn_w      = (const float*)d_in[3];
  const float* pfn_g      = (const float*)d_in[4];
  const float* pfn_b      = (const float*)d_in[5];
  // blocks: 4 + 6 + 6 conv layers, each [w,g,b] -> indices 6..53
  // deconvs: 54,57,60 ; heads: 63 cls_w, 64 cls_b, 65 box_w, 66 box_b

  struct CS { int Cin, Cout, Hi, Wi, Ho, Wo, s; };
  const CS cs[16] = {
    {64, 64,496,432,248,216,2},{64, 64,248,216,248,216,1},
    {64, 64,248,216,248,216,1},{64, 64,248,216,248,216,1},
    {64,128,248,216,124,108,2},{128,128,124,108,124,108,1},
    {128,128,124,108,124,108,1},{128,128,124,108,124,108,1},
    {128,128,124,108,124,108,1},{128,128,124,108,124,108,1},
    {128,256,124,108, 62, 54,2},{256,256, 62, 54, 62, 54,1},
    {256,256, 62, 54, 62, 54,1},{256,256, 62, 54, 62, 54,1},
    {256,256, 62, 54, 62, 54,1},{256,256, 62, 54, 62, 54,1},
  };
  const int convW[16] = {6,9,12,15, 18,21,24,27,30,33, 36,39,42,45,48,51};

  // workspace carve-up (all 256B aligned)
  char* ws = (char*)d_ws;
  size_t off = 0;
  #define WS_ALLOC(nbytes) ({ off = (off + 255) & ~(size_t)255; void* _p = ws + off; off += (size_t)(nbytes); _p; })

  const size_t canvasHp = (size_t)BC * (NYC+2) * (NXC+2) * CPFC;  // padded canvas halves
  const size_t bufHp    = (size_t)BC * (HOF+2) * (WOF+2) * 64;    // max padded act halves
  const size_t fH       = (size_t)BC * HOF * WOF * CF;            // concat halves

  float*    pillar = (float*)   WS_ALLOC((size_t)PC * CPFC * 4);
  _Float16* canvas = (_Float16*)WS_ALLOC(canvasHp * 2);
  _Float16* buf1   = (_Float16*)WS_ALLOC(bufHp * 2);
  _Float16* buf2   = (_Float16*)WS_ALLOC(bufHp * 2);
  _Float16* fbuf   = (_Float16*)WS_ALLOC(fH * 2);
  _Float16* cwp[16];
  for (int i = 0; i < 16; ++i)
    cwp[i] = (_Float16*)WS_ALLOC((size_t)cs[i].Cin * 9 * cs[i].Cout * 2);
  const int dS[3]   = {1, 2, 4};
  const int dCin[3] = {64, 128, 256};
  _Float16* dwp[3];
  for (int i = 0; i < 3; ++i)
    dwp[i] = (_Float16*)WS_ALLOC((size_t)dS[i] * dS[i] * dCin[i] * 128 * 2);
  _Float16* hwp = (_Float16*)WS_ALLOC((size_t)CF * 16 * 2);
  #undef WS_ALLOC

  // 1) weight packing (independent)
  for (int i = 0; i < 16; ++i) {
    size_t tot = (size_t)cs[i].Cin * 9 * cs[i].Cout;
    pack_conv_w<<<(unsigned)((tot + 255) / 256), 256, 0, stream>>>(
        (const float*)d_in[convW[i]], cwp[i], cs[i].Cin, cs[i].Cout);
  }
  for (int i = 0; i < 3; ++i) {
    size_t tot = (size_t)dS[i] * dS[i] * dCin[i] * 128;
    pack_deconv_w<<<(unsigned)((tot + 255) / 256), 256, 0, stream>>>(
        (const float*)d_in[54 + 3 * i], dwp[i], dCin[i], dS[i]);
  }
  pack_head_w<<<(CF * 16 + 255) / 256, 256, 0, stream>>>(
      (const float*)d_in[65], (const float*)d_in[63], hwp);

  // 2) PFN + canvas scatter (canvas fully zeroed incl. halo)
  pfn_kernel<<<PC, 64, 0, stream>>>(voxels, num_points, coors,
                                    pfn_w, pfn_g, pfn_b, pillar);
  {
    size_t zN = canvasHp / 2;  // uint count (even: CPFC is even)
    zero_u32_kernel<<<(unsigned)((zN + 255) / 256), 256, 0, stream>>>(
        (unsigned int*)canvas, zN);
  }
  scatter_kernel<<<PC, 64, 0, stream>>>(pillar, coors, canvas);

  // 3) backbone conv chain (ping-pong), halo-padded activations.
  //    Destination buffer is zeroed before each conv so its halo reads as 0.
  _Float16* cinp[16] = {canvas,buf1,buf2,buf1, buf2,buf1,buf2,buf1,buf2,buf1,
                        buf2,buf1,buf2,buf1,buf2,buf1};
  _Float16* cout_[16] = {buf1,buf2,buf1,buf2, buf1,buf2,buf1,buf2,buf1,buf2,
                         buf1,buf2,buf1,buf2,buf1,buf2};
  for (int i = 0; i < 16; ++i) {
    const int M = BC * cs[i].Ho * cs[i].Wo;
    {
      size_t zH = (size_t)BC * (cs[i].Ho + 2) * (cs[i].Wo + 2) * cs[i].Cout;
      size_t zN = zH / 2;  // uint count (Cout even)
      zero_u32_kernel<<<(unsigned)((zN + 255) / 256), 256, 0, stream>>>(
          (unsigned int*)cout_[i], zN);
    }
    dim3 g((unsigned)((M + 127) / 128), (unsigned)(cs[i].Cout / 64), 1u);
    conv3x3_wmma<<<g, 128, 0, stream>>>(
        cinp[i], cwp[i],
        (const float*)d_in[convW[i] + 1], (const float*)d_in[convW[i] + 2],
        cout_[i], BC, cs[i].Hi, cs[i].Wi, cs[i].Cin,
        cs[i].Ho, cs[i].Wo, cs[i].Cout, cs[i].s);
    if (i == 3) {   // block1 output in buf2 -> deconv0 (1x1, s=1) into f[:,0:128]
      const int Md = BC * 248 * 216;
      dim3 gd((unsigned)((Md + 127) / 128), 2u, 1u);
      deconv_wmma<<<gd, 128, 0, stream>>>(buf2, dwp[0],
          (const float*)d_in[55], (const float*)d_in[56], fbuf,
          248, 216, 64, 1, 0);
    }
    if (i == 9) {   // block2 output in buf2 -> deconv1 (2x2, s=2) into f[:,128:256]
      const int Md = BC * 124 * 108;
      dim3 gd((unsigned)((Md + 127) / 128), 2u, 4u);
      deconv_wmma<<<gd, 128, 0, stream>>>(buf2, dwp[1],
          (const float*)d_in[58], (const float*)d_in[59], fbuf,
          124, 108, 128, 2, 128);
    }
    if (i == 15) {  // block3 output in buf2 -> deconv2 (4x4, s=4) into f[:,256:384]
      const int Md = BC * 62 * 54;
      dim3 gd((unsigned)((Md + 127) / 128), 2u, 16u);
      deconv_wmma<<<gd, 128, 0, stream>>>(buf2, dwp[2],
          (const float*)d_in[61], (const float*)d_in[62], fbuf,
          62, 54, 256, 4, 256);
    }
  }

  // 4) heads: box (14) + cls (2) fused as one N=16 GEMM, fp32 out
  {
    const int M = BC * HOF * WOF;
    dim3 g((unsigned)((M + 127) / 128), 1u, 1u);
    head_wmma<<<g, 128, 0, stream>>>(fbuf, hwp,
        (const float*)d_in[66], (const float*)d_in[64], (float*)d_out);
  }
}
